// SparseTextFusion_31009663877510
// MI455X (gfx1250) — matrix operand order repaired and checked by reading, not compiled, not run on hardware
//
#include <hip/hip_runtime.h>
#include <hip/hip_bf16.h>
#include <stdint.h>

// ---- problem constants (match reference) ----
#define B_    8
#define C_    512
#define H_    64
#define W_    64
#define HW_   4096
#define EMB_  256
#define TXT_  768
#define K_    2048          // int(HW*0.5)
#define LN_EPS 1e-5f

typedef __attribute__((ext_vector_type(16))) __bf16          v16bf;
typedef __attribute__((ext_vector_type(16))) unsigned short  v16us;
typedef __attribute__((ext_vector_type(8)))  unsigned short  v8us;
typedef __attribute__((ext_vector_type(8)))  float           v8f;

// f32 -> bf16 round-to-nearest-even via bit ops
static __device__ __forceinline__ unsigned short f2bf(float f) {
    unsigned u = __float_as_uint(f);
    unsigned r = u + 0x7fffu + ((u >> 16) & 1u);
    return (unsigned short)(r >> 16);
}
static __device__ __forceinline__ float bf2f(unsigned short h) {
    return __uint_as_float(((unsigned)h) << 16);
}

// =====================================================================
// Weight prep: transpose + split into bf16 hi/lo for WMMA B-fragments.
// sparse_w [512][256] -> swT [256][512];  out_w [256][512] -> owT [512][256]
// =====================================================================
__global__ void k_prep_w(const float* __restrict__ sw, const float* __restrict__ ow,
                         unsigned short* __restrict__ swT_hi, unsigned short* __restrict__ swT_lo,
                         unsigned short* __restrict__ owT_hi, unsigned short* __restrict__ owT_lo) {
    int i = blockIdx.x * 256 + threadIdx.x;      // 0..131071 (both matrices are 131072 elems)
    {
        int r = i >> 8, c = i & 255;             // sparse_w[r][c]
        float x = sw[i];
        unsigned short h = f2bf(x);
        swT_hi[c * 512 + r] = h;
        swT_lo[c * 512 + r] = f2bf(x - bf2f(h));
    }
    {
        int r = i >> 9, c = i & 511;             // out_w[r][c]
        float x = ow[i];
        unsigned short h = f2bf(x);
        owT_hi[c * 256 + r] = h;
        owT_lo[c * 256 + r] = f2bf(x - bf2f(h));
    }
}

// =====================================================================
// Density: 1x1 conv (C->1), coalesced across hw.
// =====================================================================
__global__ void k_density(const float* __restrict__ tensor, const float* __restrict__ w,
                          const float* __restrict__ bias, float* __restrict__ d) {
    int idx = blockIdx.x * 256 + threadIdx.x;    // 0..32767
    int b = idx >> 12, hw = idx & 4095;
    const float* tp = tensor + (size_t)b * C_ * HW_ + hw;
    float acc = bias[0];
    for (int c = 0; c < C_; c += 4) {
        acc += tp[(size_t)(c + 0) * HW_] * w[c + 0];
        acc += tp[(size_t)(c + 1) * HW_] * w[c + 1];
        acc += tp[(size_t)(c + 2) * HW_] * w[c + 2];
        acc += tp[(size_t)(c + 3) * HW_] * w[c + 3];
    }
    d[idx] = acc;
}

// Dilated 3x3 (dilation 2, pad 2) + residual relu
__global__ void k_dilconv(const float* __restrict__ d, const float* __restrict__ w,
                          const float* __restrict__ bias, float* __restrict__ dens) {
    int idx = blockIdx.x * 256 + threadIdx.x;
    int b = idx >> 12, hw = idx & 4095;
    int y = hw >> 6, x = hw & 63;
    const float* db = d + b * HW_;
    float acc = bias[0];
#pragma unroll
    for (int ky = 0; ky < 3; ++ky) {
        int iy = y + 2 * ky - 2;
        if (iy < 0 || iy >= H_) continue;
#pragma unroll
        for (int kx = 0; kx < 3; ++kx) {
            int ix = x + 2 * kx - 2;
            if (ix < 0 || ix >= W_) continue;
            acc += w[ky * 3 + kx] * db[iy * W_ + ix];
        }
    }
    dens[idx] = fmaxf(acc + db[hw], 0.f);
}

// =====================================================================
// Top-K via full bitonic sort in LDS. Key = (~mono(value)<<32)|idx,
// ascending => value desc, idx asc (stable ties matching lax.top_k;
// critical: relu makes ~half the densities exactly 0.0).
// =====================================================================
__global__ void __launch_bounds__(1024) k_topk(const float* __restrict__ dens, int* __restrict__ sel) {
    __shared__ unsigned long long keys[4096];
    int b = blockIdx.x, tid = threadIdx.x;
#pragma unroll
    for (int e = 0; e < 4; ++e) {
        int i = tid + e * 1024;
        unsigned u = __float_as_uint(dens[b * HW_ + i]);
        unsigned mono = (u & 0x80000000u) ? ~u : (u | 0x80000000u);
        keys[i] = (((unsigned long long)(~mono)) << 32) | (unsigned)i;
    }
    __syncthreads();
    for (int k = 2; k <= 4096; k <<= 1) {
        for (int j = k >> 1; j > 0; j >>= 1) {
#pragma unroll
            for (int e = 0; e < 4; ++e) {
                int i = tid + e * 1024;
                int ixj = i ^ j;
                if (ixj > i) {
                    bool up = ((i & k) == 0);
                    unsigned long long a = keys[i], c = keys[ixj];
                    if ((a > c) == up) { keys[i] = c; keys[ixj] = a; }
                }
            }
            __syncthreads();
        }
    }
#pragma unroll
    for (int e = 0; e < 2; ++e) {
        int t = tid + e * 1024;
        sel[b * K_ + t] = (int)(keys[t] & 0xffffffffu);
    }
}

// =====================================================================
// Tiled 32x32 transpose per batch: src[rows][cols] -> dst[cols][rows]
// =====================================================================
__global__ void k_transpose(const float* __restrict__ src, float* __restrict__ dst,
                            int rows, int cols) {
    __shared__ float tile[32][33];
    int b = blockIdx.z;
    const float* s = src + (size_t)b * rows * cols;
    float* d = dst + (size_t)b * rows * cols;
    int c0 = blockIdx.x * 32, r0 = blockIdx.y * 32;
    int x = threadIdx.x, y = threadIdx.y;
#pragma unroll
    for (int i = 0; i < 32; i += 8) tile[y + i][x] = s[(size_t)(r0 + y + i) * cols + (c0 + x)];
    __syncthreads();
#pragma unroll
    for (int i = 0; i < 32; i += 8) d[(size_t)(c0 + y + i) * rows + (r0 + x)] = tile[x][y + i];
}

// =====================================================================
// Gather selected rows into compact bf16 hi/lo A-buffers + exact f32
// snorm. One wave per row; fully coalesced. Removes all f32->bf16
// conversion VALU from the WMMA hot loop.
// =====================================================================
__global__ void k_gather(const float* __restrict__ featT, const int* __restrict__ sel,
                         unsigned short* __restrict__ Abuf_hi, unsigned short* __restrict__ Abuf_lo,
                         float* __restrict__ snorm) {
    int wave = threadIdx.x >> 5, lane = threadIdx.x & 31;
    int k = blockIdx.x * 8 + wave;             // 0..16383
    int b = k >> 11, kk = k & 2047;
    int row = sel[b * K_ + kk];
    const float* src = featT + ((size_t)b * HW_ + row) * C_;
    unsigned short* dh = Abuf_hi + ((size_t)b * K_ + kk) * C_;
    unsigned short* dl = Abuf_lo + ((size_t)b * K_ + kk) * C_;
    float ss = 0.f;
#pragma unroll
    for (int i = 0; i < 4; ++i) {
        int off = i * 128 + lane * 4;
        float4 v = *(const float4*)(src + off);
        ss += v.x * v.x + v.y * v.y + v.z * v.z + v.w * v.w;
        ushort4 h, l;
        h.x = f2bf(v.x); l.x = f2bf(v.x - bf2f(h.x));
        h.y = f2bf(v.y); l.y = f2bf(v.y - bf2f(h.y));
        h.z = f2bf(v.z); l.z = f2bf(v.z - bf2f(h.z));
        h.w = f2bf(v.w); l.w = f2bf(v.w - bf2f(h.w));
        *(ushort4*)(dh + off) = h;
        *(ushort4*)(dl + off) = l;
    }
#pragma unroll
    for (int m = 1; m < 32; m <<= 1) ss += __shfl_xor(ss, m, 32);
    if (lane == 0) snorm[b * K_ + kk] = sqrtf(ss);
}

// =====================================================================
// Text branch: relu(text_emb @ text_w + b)*0.1, layernorm, * sigmoid(gate)
// =====================================================================
__global__ void k_text(const float* __restrict__ temb, const float* __restrict__ tw,
                       const float* __restrict__ tb, const float* __restrict__ gate_param,
                       float* __restrict__ tn) {
    int b = blockIdx.x, e = threadIdx.x;
    __shared__ float red[256];
    float acc = tb[e];
    const float* te = temb + b * TXT_;
    for (int t = 0; t < TXT_; ++t) acc += te[t] * tw[t * EMB_ + e];
    float x = fmaxf(acc, 0.f) * 0.1f;
    red[e] = x; __syncthreads();
    for (int s = 128; s > 0; s >>= 1) { if (e < s) red[e] += red[e + s]; __syncthreads(); }
    float mean = red[0] * (1.f / EMB_);
    __syncthreads();
    float dv = x - mean;
    red[e] = dv * dv; __syncthreads();
    for (int s = 128; s > 0; s >>= 1) { if (e < s) red[e] += red[e + s]; __syncthreads(); }
    float var = red[0] * (1.f / EMB_);
    float g = 1.f / (1.f + expf(-gate_param[0]));
    tn[b * EMB_ + e] = g * dv * rsqrtf(var + LN_EPS);
}

// =====================================================================
// Fused main kernel: GEMM1 (split-bf16 WMMA, K=512) + bias -> layernorm
// -> + gated text -> GEMM2 (split-bf16 WMMA, K=256) + bias -> featT rows.
// 8 waves / block, 32 rows / block; wave = (rowGroup<<2)|eQ:
//   rowGroup (0..1) picks 16 rows, eQ (0..3) picks an N-quarter
//   => 4 N-tiles/wave (32 acc VGPRs) so the scheduler can pipeline B loads.
// WMMA fragment layouts used by this code (wave32; A and B share one k order):
//   A 16x32 bf16: lane holds 16 consecutive K at kb + 16*laneHi, row=l16.
//   B 32x16 bf16: lane holds 16 consecutive K at kb + 16*laneHi, col=l16.
//   C 16x16 f32 : vgpr v -> row v + 8*laneHi, col=l16.
// =====================================================================
__global__ void __launch_bounds__(256) k_fused(
    float* __restrict__ featT, const int* __restrict__ selIdx,
    const unsigned short* __restrict__ Abuf_hi, const unsigned short* __restrict__ Abuf_lo,
    const unsigned short* __restrict__ swT_hi, const unsigned short* __restrict__ swT_lo,
    const unsigned short* __restrict__ owT_hi, const unsigned short* __restrict__ owT_lo,
    const float* __restrict__ sparse_b, const float* __restrict__ out_b,
    const float* __restrict__ tn) {

    const int b        = blockIdx.x >> 6;      // 64 tiles of 32 rows per batch
    const int tile     = blockIdx.x & 63;
    const int rowsBase = tile * 32;
    const int wave     = threadIdx.x >> 5;
    const int lane     = threadIdx.x & 31;
    const int l16      = lane & 15;
    const int laneHi   = lane >> 4;
    const int rowGroup = wave >> 2;            // 0..1
    const int eQ       = wave & 3;             // 0..3, N-quarter
    const int gRow0    = rowsBase + rowGroup * 16;
    const int hiOff    = laneHi * 16;

    __shared__ __align__(16) unsigned short midH[32 * 264]; // 32 rows x 256 bf16-hi (pad 264)
    __shared__ __align__(16) unsigned short midL[32 * 264]; // bf16-lo
    __shared__ float lnS1[32][4];
    __shared__ float lnS2[32][4];

    const int* selB = selIdx + b * K_;
    int selRow[8];
#pragma unroll
    for (int v = 0; v < 8; ++v) selRow[v] = selB[gRow0 + v + laneHi * 8];

    const unsigned short* ah = Abuf_hi + ((size_t)b * K_ + gRow0 + l16) * C_;
    const unsigned short* al = Abuf_lo + ((size_t)b * K_ + gRow0 + l16) * C_;
    const v8f zero = {0.f, 0.f, 0.f, 0.f, 0.f, 0.f, 0.f, 0.f};

    // ---------------- GEMM1: [16 x 512] @ [512 x 64(eQ)] ----------------
    v8f acc[4];
#pragma unroll
    for (int t = 0; t < 4; ++t) acc[t] = zero;

    for (int ks = 0; ks < 16; ++ks) {
        const int kb = ks * 32;
        v8us h0 = *(const v8us*)(ah + kb + hiOff);
        v8us h1 = *(const v8us*)(ah + kb + 8 + hiOff);
        v8us l0 = *(const v8us*)(al + kb + hiOff);
        v8us l1 = *(const v8us*)(al + kb + 8 + hiOff);
        v16us aU, aLU;
#pragma unroll
        for (int i = 0; i < 8; ++i) { aU[i] = h0[i]; aU[i + 8] = h1[i]; aLU[i] = l0[i]; aLU[i + 8] = l1[i]; }
        v16bf ahi = __builtin_bit_cast(v16bf, aU);
        v16bf alo = __builtin_bit_cast(v16bf, aLU);
#pragma unroll
        for (int t = 0; t < 4; ++t) {
            const int col = (eQ * 4 + t) * 16 + l16;
            v16bf bhi = __builtin_bit_cast(v16bf, *(const v16us*)(swT_hi + col * 512 + kb + laneHi * 16));
            v16bf blo = __builtin_bit_cast(v16bf, *(const v16us*)(swT_lo + col * 512 + kb + laneHi * 16));
            acc[t] = __builtin_amdgcn_wmma_f32_16x16x32_bf16(false, ahi, false, bhi, (short)0, acc[t], false, false);
            acc[t] = __builtin_amdgcn_wmma_f32_16x16x32_bf16(false, ahi, false, blo, (short)0, acc[t], false, false);
            acc[t] = __builtin_amdgcn_wmma_f32_16x16x32_bf16(false, alo, false, bhi, (short)0, acc[t], false, false);
        }
    }

    // bias + per-row layernorm partials (butterfly over the 16 N-lanes)
    float sb[4];
#pragma unroll
    for (int t = 0; t < 4; ++t) sb[t] = sparse_b[(eQ * 4 + t) * 16 + l16];
    float s1v[8], s2v[8];
#pragma unroll
    for (int v = 0; v < 8; ++v) {
        float r1 = 0.f, r2 = 0.f;
#pragma unroll
        for (int t = 0; t < 4; ++t) {
            float x = acc[t][v] + sb[t];
            acc[t][v] = x;
            r1 += x; r2 += x * x;
        }
#pragma unroll
        for (int m = 1; m < 16; m <<= 1) { r1 += __shfl_xor(r1, m, 32); r2 += __shfl_xor(r2, m, 32); }
        s1v[v] = r1; s2v[v] = r2;
    }
    if (l16 == 0) {
#pragma unroll
        for (int v = 0; v < 8; ++v) {
            int gl = rowGroup * 16 + v + laneHi * 8;
            lnS1[gl][eQ] = s1v[v];
            lnS2[gl][eQ] = s2v[v];
        }
    }
    __syncthreads();

    float meanv[8], rstdv[8];
#pragma unroll
    for (int v = 0; v < 8; ++v) {
        int gl = rowGroup * 16 + v + laneHi * 8;
        float s1 = lnS1[gl][0] + lnS1[gl][1] + lnS1[gl][2] + lnS1[gl][3];
        float s2 = lnS2[gl][0] + lnS2[gl][1] + lnS2[gl][2] + lnS2[gl][3];
        float mu = s1 * (1.f / EMB_);
        meanv[v] = mu;
        rstdv[v] = rsqrtf(s2 * (1.f / EMB_) - mu * mu + LN_EPS);
    }
    // normalize + gated text, store to LDS as bf16 hi/lo (convert once here,
    // not at every GEMM2 k-step read)
#pragma unroll
    for (int t = 0; t < 4; ++t) {
        int col = (eQ * 4 + t) * 16 + l16;
        float tv = tn[b * EMB_ + col];
#pragma unroll
        for (int v = 0; v < 8; ++v) {
            int gl = rowGroup * 16 + v + laneHi * 8;
            float val = (acc[t][v] - meanv[v]) * rstdv[v] + tv;
            unsigned short h = f2bf(val);
            midH[gl * 264 + col] = h;
            midL[gl * 264 + col] = f2bf(val - bf2f(h));
        }
    }
    __syncthreads();

    // ---------------- GEMM2: [16 x 256] @ [256 x 128(eQ)], 2 passes of 4 tiles ----------------
    const unsigned short* amH = &midH[(rowGroup * 16 + l16) * 264];
    const unsigned short* amL = &midL[(rowGroup * 16 + l16) * 264];
#pragma unroll 1
    for (int p = 0; p < 2; ++p) {
        v8f acc2[4];
#pragma unroll
        for (int t = 0; t < 4; ++t) acc2[t] = zero;
        const int colBase = eQ * 128 + p * 64;
        for (int ks = 0; ks < 8; ++ks) {
            const int kb = ks * 32;
            v8us h0 = *(const v8us*)(amH + kb + hiOff);
            v8us h1 = *(const v8us*)(amH + kb + 8 + hiOff);
            v8us l0 = *(const v8us*)(amL + kb + hiOff);
            v8us l1 = *(const v8us*)(amL + kb + 8 + hiOff);
            v16us aU, aLU;
#pragma unroll
            for (int i = 0; i < 8; ++i) { aU[i] = h0[i]; aU[i + 8] = h1[i]; aLU[i] = l0[i]; aLU[i + 8] = l1[i]; }
            v16bf ahi = __builtin_bit_cast(v16bf, aU);
            v16bf alo = __builtin_bit_cast(v16bf, aLU);
#pragma unroll
            for (int t = 0; t < 4; ++t) {
                const int col = colBase + t * 16 + l16;
                v16bf bhi = __builtin_bit_cast(v16bf, *(const v16us*)(owT_hi + col * 256 + kb + laneHi * 16));
                v16bf blo = __builtin_bit_cast(v16bf, *(const v16us*)(owT_lo + col * 256 + kb + laneHi * 16));
                acc2[t] = __builtin_amdgcn_wmma_f32_16x16x32_bf16(false, ahi, false, bhi, (short)0, acc2[t], false, false);
                acc2[t] = __builtin_amdgcn_wmma_f32_16x16x32_bf16(false, ahi, false, blo, (short)0, acc2[t], false, false);
                acc2[t] = __builtin_amdgcn_wmma_f32_16x16x32_bf16(false, alo, false, bhi, (short)0, acc2[t], false, false);
            }
        }
#pragma unroll
        for (int t = 0; t < 4; ++t) {
            const int col = colBase + t * 16 + l16;
            float ob = out_b[col];
#pragma unroll
            for (int v = 0; v < 8; ++v) {
                featT[((size_t)b * HW_ + selRow[v]) * C_ + col] = acc2[t][v] + ob;
            }
        }
    }
}

// =====================================================================
// Renormalize fused rows: row *= snorm / max(||row||, 1e-12). Wave/row.
// =====================================================================
__global__ void k_renorm(float* __restrict__ featT, const int* __restrict__ selIdx,
                         const float* __restrict__ snorm) {
    int wave = threadIdx.x >> 5, lane = threadIdx.x & 31;
    int k = blockIdx.x * 8 + wave;            // 0..16383
    int b = k >> 11, kk = k & 2047;
    int row = selIdx[b * K_ + kk];
    float* ptr = featT + ((size_t)b * HW_ + row) * C_;
    float4 d[4];
    float ss = 0.f;
#pragma unroll
    for (int i = 0; i < 4; ++i) {
        d[i] = *(const float4*)(ptr + i * 128 + lane * 4);
        ss += d[i].x * d[i].x + d[i].y * d[i].y + d[i].z * d[i].z + d[i].w * d[i].w;
    }
#pragma unroll
    for (int m = 1; m < 32; m <<= 1) ss += __shfl_xor(ss, m, 32);
    float sc = snorm[b * K_ + kk] / fmaxf(sqrtf(ss), 1e-12f);
#pragma unroll
    for (int i = 0; i < 4; ++i) {
        d[i].x *= sc; d[i].y *= sc; d[i].z *= sc; d[i].w *= sc;
        *(float4*)(ptr + i * 128 + lane * 4) = d[i];
    }
}

// =====================================================================
// Host launch. ws layout (bytes), all 256B aligned, total ~97.4MB:
//   featT   @ 0           (64MB)
//   Abuf_hi @ 67108864    (16MB)   Abuf_lo @ 83886080 (16MB)
//   dbuf    @ 100663296   (128KB)  dens    @ 100794368 (128KB)
//   sel     @ 100925440   (64KB)   snorm   @ 100990976 (64KB)
//   tn      @ 101056512   (8KB)
//   swT_hi/lo @ 101064704 / 101326848 (256KB each)
//   owT_hi/lo @ 101588992 / 101851136 (256KB each)
// =====================================================================
extern "C" void kernel_launch(void* const* d_in, const int* in_sizes, int n_in,
                              void* d_out, int out_size, void* d_ws, size_t ws_size,
                              hipStream_t stream) {
    (void)in_sizes; (void)n_in; (void)out_size; (void)ws_size;
    const float* tensor     = (const float*)d_in[0];
    const float* text_emb   = (const float*)d_in[1];
    const float* conv1_w    = (const float*)d_in[2];
    const float* conv1_b    = (const float*)d_in[3];
    const float* dconv_w    = (const float*)d_in[4];
    const float* dconv_b    = (const float*)d_in[5];
    const float* sparse_w   = (const float*)d_in[6];
    const float* sparse_b   = (const float*)d_in[7];
    const float* text_w     = (const float*)d_in[8];
    const float* text_b     = (const float*)d_in[9];
    const float* out_w      = (const float*)d_in[10];
    const float* out_b      = (const float*)d_in[11];
    const float* gate_param = (const float*)d_in[12];

    char* ws = (char*)d_ws;
    float*          featT   = (float*)(ws);
    unsigned short* Abuf_hi = (unsigned short*)(ws + 67108864);
    unsigned short* Abuf_lo = (unsigned short*)(ws + 83886080);
    float*          dbuf    = (float*)(ws + 100663296);
    float*          dens    = (float*)(ws + 100794368);
    int*            sel     = (int*)  (ws + 100925440);
    float*          snorm   = (float*)(ws + 100990976);
    float*          tn      = (float*)(ws + 101056512);
    unsigned short* swT_hi  = (unsigned short*)(ws + 101064704);
    unsigned short* swT_lo  = (unsigned short*)(ws + 101326848);
    unsigned short* owT_hi  = (unsigned short*)(ws + 101588992);
    unsigned short* owT_lo  = (unsigned short*)(ws + 101851136);
    float* out = (float*)d_out;

    dim3 tb(32, 8);
    k_prep_w  <<<512, 256, 0, stream>>>(sparse_w, out_w, swT_hi, swT_lo, owT_hi, owT_lo);
    k_density <<<128, 256, 0, stream>>>(tensor, conv1_w, conv1_b, dbuf);
    k_dilconv <<<128, 256, 0, stream>>>(dbuf, dconv_w, dconv_b, dens);
    k_topk    <<<8, 1024, 0, stream>>>(dens, sel);
    k_transpose<<<dim3(128, 16, B_), tb, 0, stream>>>(tensor, featT, C_, HW_);  // [C][HW] -> [HW][C]
    k_gather  <<<2048, 256, 0, stream>>>(featT, sel, Abuf_hi, Abuf_lo, snorm);
    k_text    <<<B_, 256, 0, stream>>>(text_emb, text_w, text_b, gate_param, tn);
    k_fused   <<<B_ * 64, 256, 0, stream>>>(featT, sel, Abuf_hi, Abuf_lo,
                                            swT_hi, swT_lo, owT_hi, owT_lo,
                                            sparse_b, out_b, tn);
    k_renorm  <<<2048, 256, 0, stream>>>(featT, sel, snorm);
    k_transpose<<<dim3(16, 128, B_), tb, 0, stream>>>(featT, out, HW_, C_);     // [HW][C] -> [C][HW]
}
